// Attention_18811956757096
// MI455X (gfx1250) — matrix operand order repaired and checked by reading, not compiled, not run on hardware
//
#include <hip/hip_runtime.h>
#include <hip/hip_bf16.h>

// ---------------------------------------------------------------------------
// CDNA5 (gfx1250) attention block, bf16 WMMA path + async global->LDS staging.
//   B=2, T=2048, D=1024, H=16, hd=64
// ---------------------------------------------------------------------------

typedef __attribute__((ext_vector_type(16))) __bf16 v16bf;
typedef __attribute__((ext_vector_type(8)))  __bf16 v8bf;
typedef __attribute__((ext_vector_type(8)))  float  v8f;

static __device__ __forceinline__ unsigned short f2bf(float f) {
    unsigned int u = __float_as_uint(f);
    u += 0x7FFFu + ((u >> 16) & 1u);          // round-to-nearest-even
    return (unsigned short)(u >> 16);
}

// Build a v16bf A/B fragment from two 16-byte-aligned 8-element chunks.
static __device__ __forceinline__ v16bf ld_frag16(const unsigned short* p0,
                                                  const unsigned short* p1) {
    v8bf lo = *reinterpret_cast<const v8bf*>(p0);
    v8bf hi = *reinterpret_cast<const v8bf*>(p1);
    v16bf r;
#pragma unroll
    for (int i = 0; i < 8; ++i) { r[i] = lo[i]; r[i + 8] = hi[i]; }
    return r;
}

// ---------------------------------------------------------------------------
// Async global->LDS (CDNA5 GLOBAL_LOAD_ASYNC_TO_LDS_B128, ASYNCcnt-tracked).
// Builtin prototype (from probe): (int4 AS1*, int4 AS3*, imm offset, imm cpol)
// ---------------------------------------------------------------------------
#if defined(__gfx1250__) && __has_builtin(__builtin_amdgcn_global_load_async_to_lds_b128)
#define HAVE_ASYNC_LDS 1
#endif

#if defined(HAVE_ASYNC_LDS)
typedef int v4i_t __attribute__((vector_size(16)));
typedef __attribute__((address_space(1))) v4i_t as1_v4i;
typedef __attribute__((address_space(3))) v4i_t as3_v4i;
static __device__ __forceinline__ void stage16(const unsigned short* g,
                                               unsigned short* l) {
    __builtin_amdgcn_global_load_async_to_lds_b128(
        (as1_v4i*)(void*)g, (as3_v4i*)(void*)l, 0, 0);
}
static __device__ __forceinline__ void stage_wait() {
#if __has_builtin(__builtin_amdgcn_s_wait_asynccnt)
    __builtin_amdgcn_s_wait_asynccnt(0);
#else
    asm volatile("s_wait_asynccnt 0x0" ::: "memory");
#endif
}
#else
static __device__ __forceinline__ void stage16(const unsigned short* g,
                                               unsigned short* l) {
    *reinterpret_cast<uint4*>(l) = *reinterpret_cast<const uint4*>(g);
}
static __device__ __forceinline__ void stage_wait() {}
#endif

// ---------------------------------------------------------------------------
// 1) f32 -> bf16 bulk conversion (4 elements / thread)
// ---------------------------------------------------------------------------
__global__ void cvt_f32_bf16_kernel(const float* __restrict__ src,
                                    unsigned short* __restrict__ dst, int n4) {
    int i = blockIdx.x * blockDim.x + threadIdx.x;
    if (i >= n4) return;
    float4 v = reinterpret_cast<const float4*>(src)[i];
    unsigned int a = (unsigned int)f2bf(v.x) | ((unsigned int)f2bf(v.y) << 16);
    unsigned int b = (unsigned int)f2bf(v.z) | ((unsigned int)f2bf(v.w) << 16);
    reinterpret_cast<uint2*>(dst)[i] = make_uint2(a, b);
}

// ---------------------------------------------------------------------------
// 2) bf16 GEMM: C[M,N] f32 = A[M,K]bf16 @ B[K,N]bf16 + bias[N]
//    128x128 block tile, K-step 32, double-buffered async LDS staging,
//    8 waves x 8 WMMAs per K-step, one barrier per K-step.
// ---------------------------------------------------------------------------
#define ALD 40   // 32 + 8 pad (conflict-free ds_read_b128 pattern)
#define BLD 136  // 128 + 8 pad

__global__ __launch_bounds__(256)
void gemm_bf16_kernel(const unsigned short* __restrict__ A,
                      const unsigned short* __restrict__ B,
                      const float* __restrict__ bias,
                      float* __restrict__ C, int M, int N, int K) {
    __shared__ __align__(16) unsigned short As[2][128 * ALD];
    __shared__ __align__(16) unsigned short Bs[2][32 * BLD];

    const int tid  = threadIdx.x;
    const int m0   = blockIdx.y * 128;
    const int n0   = blockIdx.x * 128;
    const int w    = tid >> 5, lane = tid & 31;
    const int wrow = w >> 1,  wcol = w & 1;
    const int lm   = lane & 15;
    const int co   = (lane < 16) ? 0 : 8;     // A-layout chunk offset

    const int ar = tid >> 1, ah = tid & 1;    // A tile 128x32: 32B / thread
    const int br = tid >> 3, bc = tid & 7;    // B tile 32x128: 32B / thread

    v8f acc[2][4];
#pragma unroll
    for (int mi = 0; mi < 2; ++mi)
#pragma unroll
        for (int ni = 0; ni < 4; ++ni) acc[mi][ni] = {};

    // stage K-step 0 into buffer 0
    {
        const unsigned short* gp = A + (size_t)(m0 + ar) * K + ah * 16;
        unsigned short* sp = &As[0][ar * ALD + ah * 16];
        stage16(gp, sp);
        stage16(gp + 8, sp + 8);
        const unsigned short* gq = B + (size_t)br * N + n0 + bc * 16;
        unsigned short* sq = &Bs[0][br * BLD + bc * 16];
        stage16(gq, sq);
        stage16(gq + 8, sq + 8);
    }

    int buf = 0;
    for (int k0 = 0; k0 < K; k0 += 32) {
        stage_wait();          // own stage(k0) complete
        __syncthreads();       // everyone's stage(k0) complete; buf^1 free
        if (k0 + 32 < K) {
            const unsigned short* gp = A + (size_t)(m0 + ar) * K + k0 + 32 + ah * 16;
            unsigned short* sp = &As[buf ^ 1][ar * ALD + ah * 16];
            stage16(gp, sp);
            stage16(gp + 8, sp + 8);
            const unsigned short* gq = B + (size_t)(k0 + 32 + br) * N + n0 + bc * 16;
            unsigned short* sq = &Bs[buf ^ 1][br * BLD + bc * 16];
            stage16(gq, sq);
            stage16(gq + 8, sq + 8);
        }

        v16bf afr[2], bfr[4];
#pragma unroll
        for (int mi = 0; mi < 2; ++mi) {
            const unsigned short* ap = &As[buf][(wrow * 32 + mi * 16 + lm) * ALD];
            afr[mi] = ld_frag16(ap + co, ap + 16 + co);
        }
#pragma unroll
        for (int ni = 0; ni < 4; ++ni) {
            const unsigned short* bp = &Bs[buf][wcol * 64 + ni * 16 + lm];
#pragma unroll
            for (int s = 0; s < 16; ++s)
                bfr[ni][s] = __builtin_bit_cast(__bf16, bp[((s < 8 ? s : s + 8) + co) * BLD]);
        }
#pragma unroll
        for (int mi = 0; mi < 2; ++mi)
#pragma unroll
            for (int ni = 0; ni < 4; ++ni)
                acc[mi][ni] = __builtin_amdgcn_wmma_f32_16x16x32_bf16(
                    false, afr[mi], false, bfr[ni], (short)0, acc[mi][ni],
                    false, false);
        buf ^= 1;
    }

    // epilogue: C-layout -> global, add bias
#pragma unroll
    for (int mi = 0; mi < 2; ++mi)
#pragma unroll
        for (int ni = 0; ni < 4; ++ni) {
            int col = n0 + wcol * 64 + ni * 16 + lm;
            float bv = bias[col];
#pragma unroll
            for (int r = 0; r < 8; ++r) {
                int row = m0 + wrow * 32 + mi * 16 + r + co;
                C[(size_t)row * N + col] = acc[mi][ni][r] + bv;
            }
        }
}

// ---------------------------------------------------------------------------
// 3) RoPE + split + relayout. qkv f32 [B*T, 3*1024] ->
//    qb [b,h,t,d] bf16 (scaled by 1/8 * log2(e)),
//    kt [b,h,d,t] bf16 (transposed for contiguous k^T staging),
//    vb [b,h,t,d] bf16.  One thread per (b,h,t,d<32).
// ---------------------------------------------------------------------------
__global__ void rope_split_kernel(const float* __restrict__ qkv,
                                  unsigned short* __restrict__ qb,
                                  unsigned short* __restrict__ kt,
                                  unsigned short* __restrict__ vb) {
    const int T = 2048, H = 16;
    int idx = blockIdx.x * blockDim.x + threadIdx.x;   // < 2*16*2048*32
    int d = idx & 31;
    int t = (idx >> 5) & 2047;
    int h = (idx >> 16) & 15;
    int b = idx >> 20;

    size_t row = (size_t)(b * T + t) * 3072;
    int cq = h * 64 + d;
    float q1 = qkv[row + cq],        q2 = qkv[row + cq + 32];
    float k1 = qkv[row + 1024 + cq], k2 = qkv[row + 1024 + cq + 32];
    float v1 = qkv[row + 2048 + cq], v2 = qkv[row + 2048 + cq + 32];

    // inv_freq = 10000^(-d/32)
    float inv_freq = exp2f(-(float)d * (13.287712379549449f / 32.0f));
    float fr = (float)t * inv_freq;
    float s, c;
    __sincosf(fr, &s, &c);

    const float QS = 0.125f * 1.44269504088896f;  // 1/sqrt(hd) * log2(e)
    float qr1 = (q1 * c - q2 * s) * QS;
    float qr2 = (q1 * s + q2 * c) * QS;
    float kr1 = k1 * c - k2 * s;
    float kr2 = k1 * s + k2 * c;

    size_t bh = (size_t)(b * H + h);
    size_t qv = (bh * T + t) * 64 + d;
    qb[qv]      = f2bf(qr1);
    qb[qv + 32] = f2bf(qr2);
    vb[qv]      = f2bf(v1);
    vb[qv + 32] = f2bf(v2);
    size_t kb = (bh * 64 + d) * T + t;
    kt[kb]                  = f2bf(kr1);
    kt[kb + 32 * (size_t)T] = f2bf(kr2);
}

// ---------------------------------------------------------------------------
// 4) Flash attention. Block = 8 waves sharing one (b,h); wave owns a 16-row
//    q tile. 32 keys / iteration. K^T (64x32) and V (32x64) tiles are staged
//    cooperatively into double-buffered LDS (8x data reuse across waves).
//    scores: 4 WMMAs; online softmax via wave32 shuffles; probs transposed
//    to A-layout through per-wave LDS; PV: 4 WMMAs.
// ---------------------------------------------------------------------------
#define KLD 40   // 32 + 8 pad
#define VLD 72   // 64 + 8 pad

__global__ __launch_bounds__(256)
void flash_attn_kernel(const unsigned short* __restrict__ qb,
                       const unsigned short* __restrict__ ktr,
                       const unsigned short* __restrict__ vb,
                       unsigned short* __restrict__ ao) {
    const int T = 2048;
    __shared__ __align__(16) unsigned short Ks[2][64 * KLD];
    __shared__ __align__(16) unsigned short Vs[2][32 * VLD];
    __shared__ __align__(16) unsigned short Ps[8][16 * 32];

    const int tid = threadIdx.x;
    const int w = tid >> 5, lane = tid & 31;
    const int lm = lane & 15;
    const int co = (lane < 16) ? 0 : 8;
    const int bh = blockIdx.y;                 // b*16 + h
    const int b  = bh >> 4, h = bh & 15;
    const int t0 = (blockIdx.x * 8 + w) * 16;
    const size_t bhT = (size_t)bh * T;
    const unsigned short* ktb = ktr + (size_t)bh * 64 * T;

    const int krow = tid >> 2, kc4 = tid & 3;  // K tile 64x32: 16B / thread
    const int vrw  = tid >> 3, vc8 = tid & 7;  // V tile 32x64: 16B / thread

    // persistent q A-fragments: 16x64 as two K=32 chunks
    const unsigned short* qrow = qb + (bhT + t0 + lm) * 64;
    v16bf aq0 = ld_frag16(qrow + co,      qrow + 16 + co);
    v16bf aq1 = ld_frag16(qrow + 32 + co, qrow + 48 + co);

    v8f o0 = {}, o1 = {}, o2 = {}, o3 = {};
    float run_max[8], run_sum[8];
#pragma unroll
    for (int r = 0; r < 8; ++r) { run_max[r] = -3.0e38f; run_sum[r] = 0.f; }
    unsigned short* pw = &Ps[w][0];

    // stage iteration 0 into buffer 0
    stage16(ktb + (size_t)krow * T + kc4 * 8, &Ks[0][krow * KLD + kc4 * 8]);
    stage16(vb + (bhT + vrw) * 64 + vc8 * 8,  &Vs[0][vrw * VLD + vc8 * 8]);

    int buf = 0;
    for (int j = 0; j < T / 32; ++j) {
        stage_wait();
        __syncthreads();
        if (j + 1 < T / 32) {
            int kb2 = (j + 1) * 32;
            stage16(ktb + (size_t)krow * T + kb2 + kc4 * 8,
                    &Ks[buf ^ 1][krow * KLD + kc4 * 8]);
            stage16(vb + (bhT + kb2 + vrw) * 64 + vc8 * 8,
                    &Vs[buf ^ 1][vrw * VLD + vc8 * 8]);
        }

        // k^T B-fragments from LDS: lane = key column, elements = head-dim (table order)
        v16bf b00, b10, b01, b11;
#pragma unroll
        for (int s = 0; s < 16; ++s) {
            int kd = (s < 8 ? s : s + 8) + co;
            b00[s] = __builtin_bit_cast(__bf16, Ks[buf][kd * KLD + lm]);
            b10[s] = __builtin_bit_cast(__bf16, Ks[buf][kd * KLD + 16 + lm]);
            b01[s] = __builtin_bit_cast(__bf16, Ks[buf][(32 + kd) * KLD + lm]);
            b11[s] = __builtin_bit_cast(__bf16, Ks[buf][(32 + kd) * KLD + 16 + lm]);
        }

        v8f z = {};
        v8f s0 = __builtin_amdgcn_wmma_f32_16x16x32_bf16(false, aq0, false, b00,
                                                         (short)0, z, false, false);
        s0 = __builtin_amdgcn_wmma_f32_16x16x32_bf16(false, aq1, false, b01,
                                                     (short)0, s0, false, false);
        v8f s1 = __builtin_amdgcn_wmma_f32_16x16x32_bf16(false, aq0, false, b10,
                                                         (short)0, z, false, false);
        s1 = __builtin_amdgcn_wmma_f32_16x16x32_bf16(false, aq1, false, b11,
                                                     (short)0, s1, false, false);

        // online softmax per row r (+8 for upper half-wave)
#pragma unroll
        for (int r = 0; r < 8; ++r) {
            float vm = fmaxf(s0[r], s1[r]);
            vm = fmaxf(vm, __shfl_xor(vm, 1, 32));
            vm = fmaxf(vm, __shfl_xor(vm, 2, 32));
            vm = fmaxf(vm, __shfl_xor(vm, 4, 32));
            vm = fmaxf(vm, __shfl_xor(vm, 8, 32));
            float nm = fmaxf(run_max[r], vm);
            float cf = exp2f(run_max[r] - nm);
            run_max[r] = nm;
            float p0 = exp2f(s0[r] - nm);
            float p1 = exp2f(s1[r] - nm);
            float rs = p0 + p1;
            rs += __shfl_xor(rs, 1, 32);
            rs += __shfl_xor(rs, 2, 32);
            rs += __shfl_xor(rs, 4, 32);
            rs += __shfl_xor(rs, 8, 32);
            run_sum[r] = run_sum[r] * cf + rs;
            o0[r] *= cf; o1[r] *= cf; o2[r] *= cf; o3[r] *= cf;
            int m = r + co;
            pw[m * 32 + lm]      = f2bf(p0);   // C-layout -> LDS row-major
            pw[m * 32 + 16 + lm] = f2bf(p1);
        }

        // probs back as 16x32 A-fragment (per-wave LDS, same-wave ordering)
        const unsigned short* pr = pw + lm * 32;
        v16bf pa = ld_frag16(pr + co, pr + 16 + co);

        // V B-fragments from LDS: lane = head-dim column, elements = keys (table order)
        v16bf bv0, bv1, bv2, bv3;
#pragma unroll
        for (int s = 0; s < 16; ++s) {
            const unsigned short* vl = &Vs[buf][((s < 8 ? s : s + 8) + co) * VLD + lm];
            bv0[s] = __builtin_bit_cast(__bf16, vl[0]);
            bv1[s] = __builtin_bit_cast(__bf16, vl[16]);
            bv2[s] = __builtin_bit_cast(__bf16, vl[32]);
            bv3[s] = __builtin_bit_cast(__bf16, vl[48]);
        }
        o0 = __builtin_amdgcn_wmma_f32_16x16x32_bf16(false, pa, false, bv0,
                                                     (short)0, o0, false, false);
        o1 = __builtin_amdgcn_wmma_f32_16x16x32_bf16(false, pa, false, bv1,
                                                     (short)0, o1, false, false);
        o2 = __builtin_amdgcn_wmma_f32_16x16x32_bf16(false, pa, false, bv2,
                                                     (short)0, o2, false, false);
        o3 = __builtin_amdgcn_wmma_f32_16x16x32_bf16(false, pa, false, bv3,
                                                     (short)0, o3, false, false);
        buf ^= 1;
    }

    // finalize: divide by softmax denominator, write bf16 [B,T,D]
#pragma unroll
    for (int r = 0; r < 8; ++r) {
        float inv = 1.0f / run_sum[r];
        int t = t0 + r + co;
        size_t orow = ((size_t)b * T + t) * 1024 + h * 64 + lm;
        ao[orow]      = f2bf(o0[r] * inv);
        ao[orow + 16] = f2bf(o1[r] * inv);
        ao[orow + 32] = f2bf(o2[r] * inv);
        ao[orow + 48] = f2bf(o3[r] * inv);
    }
}

// ---------------------------------------------------------------------------
// Host launch
// ---------------------------------------------------------------------------
extern "C" void kernel_launch(void* const* d_in, const int* in_sizes, int n_in,
                              void* d_out, int out_size, void* d_ws, size_t ws_size,
                              hipStream_t stream) {
    const float* x     = (const float*)d_in[0];  // [2,2048,1024]
    const float* w_qkv = (const float*)d_in[1];  // [1024,3072]
    const float* b_qkv = (const float*)d_in[2];  // [3072]
    const float* w_out = (const float*)d_in[3];  // [1024,1024]
    const float* b_out = (const float*)d_in[4];  // [1024]
    float* out = (float*)d_out;

    char* ws = (char*)d_ws;
    unsigned short* XB    = (unsigned short*)(ws + 0);          //  8.0 MB x bf16
    unsigned short* WQKVB = (unsigned short*)(ws + 8388608);    //  6.0 MB
    unsigned short* WOUTB = (unsigned short*)(ws + 14680064);   //  2.0 MB
    float*          QKVF  = (float*)        (ws + 16777216);    // 48.0 MB qkv f32
    unsigned short* QB    = (unsigned short*)(ws + 67108864);   //  8.0 MB
    unsigned short* KT    = (unsigned short*)(ws + 75497472);   //  8.0 MB (k^T)
    unsigned short* VB    = (unsigned short*)(ws + 83886080);   //  8.0 MB
    unsigned short* AOB   = (unsigned short*)(ws + 92274688);   //  8.0 MB attn out

    // 1) convert inputs/weights to bf16
    cvt_f32_bf16_kernel<<<4096, 256, 0, stream>>>(x,     XB,    4194304 / 4);
    cvt_f32_bf16_kernel<<<3072, 256, 0, stream>>>(w_qkv, WQKVB, 3145728 / 4);
    cvt_f32_bf16_kernel<<<1024, 256, 0, stream>>>(w_out, WOUTB, 1048576 / 4);

    // 2) QKV projection: [4096,1024] @ [1024,3072] + b_qkv
    gemm_bf16_kernel<<<dim3(3072 / 128, 4096 / 128), 256, 0, stream>>>(
        XB, WQKVB, b_qkv, QKVF, 4096, 3072, 1024);

    // 3) RoPE + split + relayout
    rope_split_kernel<<<8192, 256, 0, stream>>>(QKVF, QB, KT, VB);

    // 4) flash attention over 32 (b,h) pairs, 128 q-tiles each
    flash_attn_kernel<<<dim3(16, 32), 256, 0, stream>>>(QB, KT, VB, AOB);

    // 5) output projection: [4096,1024] @ [1024,1024] + b_out -> d_out
    gemm_bf16_kernel<<<dim3(1024 / 128, 4096 / 128), 256, 0, stream>>>(
        AOB, WOUTB, b_out, out, 4096, 1024, 1024);
}
